// GIB_41016937677241
// MI455X (gfx1250) — matrix operand order repaired and checked by reading, not compiled, not run on hardware
//
#include <hip/hip_runtime.h>

#define NUM_GRAPHS 2048
#define NPG 64          // nodes per graph
#define DEG 16
#define DD 256          // feature dim
#define DH 128          // hidden dim
#define NN (NUM_GRAPHS * NPG)
#define EE (NN * DEG)

typedef __attribute__((ext_vector_type(16))) __bf16 v16bf;
typedef __attribute__((ext_vector_type(8)))  float  v8f;

// One workgroup (128 threads = 4 waves) per graph.
// Wave w computes rows [g*64 + 16w, +16) of a = softmax(tanh(h w1^T + b1) w2^T + b2),
// then the block does the per-graph reductions using a held in LDS.
__global__ __launch_bounds__(128) void fused_graph_kernel(
    const float* __restrict__ h,
    const int*   __restrict__ esrc,
    const int*   __restrict__ edst,
    const float* __restrict__ w1,
    const float* __restrict__ b1,
    const float* __restrict__ w2,
    const float* __restrict__ b2,
    float* __restrict__ out_h,
    float* __restrict__ ge,
    float* __restrict__ pos,
    float* __restrict__ pen)
{
    // 64KB LDS: holds bf16 B-fragments of w1^T during the GEMM, then is
    // reused for a[64] (float2) and the edge reduction buffer.
    __shared__ __align__(32) char smem[65536];
    __bf16* sB  = (__bf16*)smem;
    float2* sa  = (float2*)smem;           // 64 * 8B   = 512B  (reuse)
    float4* red = (float4*)(smem + 1024);  // 128 * 16B = 2KB   (reuse)

    const int g     = blockIdx.x;
    const int tid   = threadIdx.x;
    const int wave  = tid >> 5;
    const int lane  = tid & 31;
    const int lm    = lane & 15;
    const int khalf = lane >> 4;

    // ---- Phase 1: stage w1^T (K=256 x N=128) as bf16 WMMA B-fragments ----
    // B 32x16 (16-bit) layout: lane l -> col n = l&15, K = 8*(l>>4) + (e<8 ? e : e+8)
    // flat idx = (((ks*8)+nt)*32 + l)*16 + e
    for (int idx = tid; idx < 8 * 8 * 32 * 16; idx += 128) {
        int e  = idx & 15;
        int l  = (idx >> 4) & 31;
        int nt = (idx >> 9) & 7;
        int ks = idx >> 12;
        int kk = ks * 32 + 8 * (l >> 4) + (e < 8 ? e : e + 8);
        int nn = nt * 16 + (l & 15);
        sB[idx] = (__bf16)w1[nn * DD + kk];
    }
    __syncthreads();

    // ---- Phase 2: 16x128 GEMM tile per wave via v_wmma_f32_16x16x32_bf16 ----
    const int    row0 = g * NPG + wave * 16;
    const float* hrow = h + (size_t)(row0 + lm) * DD;

    v8f acc[8];
#pragma unroll
    for (int j = 0; j < 8; ++j) acc[j] = (v8f){0.f,0.f,0.f,0.f,0.f,0.f,0.f,0.f};

    for (int ks = 0; ks < 8; ++ks) {
        // A 16x32 (16-bit) layout: lane row m = lane&15;
        // lanes 0-15: K = {0..7, 16..23}; lanes 16-31: K = {8..15, 24..31}
        v16bf afrag;
        const float* ap = hrow + ks * 32 + 8 * khalf;
#pragma unroll
        for (int e = 0; e < 8; ++e) {
            afrag[e]     = (__bf16)ap[e];
            afrag[e + 8] = (__bf16)ap[e + 16];
        }
#pragma unroll
        for (int j = 0; j < 8; ++j) {
            const v16bf bfrag = *(const v16bf*)(sB + (size_t)((ks * 8 + j) * 32 + lane) * 16);
            acc[j] = __builtin_amdgcn_wmma_f32_16x16x32_bf16(
                false, afrag, false, bfrag, (short)0, acc[j], false, false);
        }
    }

    // acc[j][r] holds C at row m = r + 8*khalf, col n = 16*j + lm.
    float b1j[8], w20[8], w21[8];
#pragma unroll
    for (int j = 0; j < 8; ++j) {
        b1j[j] = b1[j * 16 + lm];
        w20[j] = w2[j * 16 + lm];
        w21[j] = w2[DH + j * 16 + lm];
    }
#pragma unroll
    for (int j = 0; j < 8; ++j)
#pragma unroll
        for (int r = 0; r < 8; ++r)
            acc[j][r] = tanhf(acc[j][r] + b1j[j]);

    // 128 -> 2 projection + softmax (in-register dot + xor-shuffle over 16 lanes)
    const float b20 = b2[0], b21 = b2[1];
    float myA0 = 0.f, myA1 = 0.f;
    int   myRow = -1;
#pragma unroll
    for (int r = 0; r < 8; ++r) {
        float s0 = 0.f, s1 = 0.f;
#pragma unroll
        for (int j = 0; j < 8; ++j) {
            s0 += acc[j][r] * w20[j];
            s1 += acc[j][r] * w21[j];
        }
#pragma unroll
        for (int m = 1; m <= 8; m <<= 1) {
            s0 += __shfl_xor(s0, m, 32);
            s1 += __shfl_xor(s1, m, 32);
        }
        if (lm == r) {
            float z0 = s0 + b20, z1 = s1 + b21;
            float mx = fmaxf(z0, z1);
            float e0 = __expf(z0 - mx), e1 = __expf(z1 - mx);
            float inv = 1.0f / (e0 + e1);
            myA0 = e0 * inv;
            myA1 = e1 * inv;
            myRow = wave * 16 + r + 8 * khalf;
        }
    }

    __syncthreads();                       // everyone done reading sB
    if (myRow >= 0) sa[myRow] = make_float2(myA0, myA1);
    __syncthreads();

    // ---- Phase 3: h copy-through + a-weighted per-graph sums (coalesced) ----
    {
        const size_t base = (size_t)g * NPG * DD;
        float s0a = 0.f, s1a = 0.f, s0b = 0.f, s1b = 0.f;
        for (int n = 0; n < NPG; ++n) {
            float2 an = sa[n];
            size_t ia = base + (size_t)n * DD + tid;
            float hva = h[ia];
            float hvb = h[ia + 128];
            out_h[ia]       = hva;
            out_h[ia + 128] = hvb;
            s0a += an.x * hva; s1a += an.y * hva;
            s0b += an.x * hvb; s1b += an.y * hvb;
        }
        ge[(size_t)g * DD + tid]        = 0.5f * (s0a + s1a);
        ge[(size_t)g * DD + tid + 128]  = 0.5f * (s0b + s1b);
        pos[(size_t)g * DD + tid]       = s0a;
        pos[(size_t)g * DD + tid + 128] = s0b;
    }

    // ---- Phase 4: 2x2 edge adjacency sums -> L1-normalized diag penalty ----
    {
        float m00 = 0.f, m01 = 0.f, m10 = 0.f, m11 = 0.f;
        const int e0 = g * NPG * DEG;      // 1024 edges per graph, contiguous
#pragma unroll
        for (int k = 0; k < 8; ++k) {
            int e  = e0 + tid + 128 * k;
            int sl = esrc[e] & (NPG - 1);
            int dl = edst[e] & (NPG - 1);
            float2 as = sa[sl];
            float2 ad = sa[dl];
            m00 += as.x * ad.x; m01 += as.x * ad.y;
            m10 += as.y * ad.x; m11 += as.y * ad.y;
        }
        red[tid] = make_float4(m00, m01, m10, m11);
        __syncthreads();
        for (int off = 64; off > 0; off >>= 1) {
            if (tid < off) {
                float4 o = red[tid + off];
                float4 m = red[tid];
                m.x += o.x; m.y += o.y; m.z += o.z; m.w += o.w;
                red[tid] = m;
            }
            __syncthreads();
        }
        if (tid == 0) {
            float4 M = red[0];
            float r0 = fmaxf(fabsf(M.x) + fabsf(M.y), 1e-12f);
            float r1 = fmaxf(fabsf(M.z) + fabsf(M.w), 1e-12f);
            float d0 = M.x / r0 - 1.0f;
            float d1 = M.w / r1 - 1.0f;
            pen[g] = 0.5f * (d0 * d0 + d1 * d1);
        }
    }
}

extern "C" void kernel_launch(void* const* d_in, const int* in_sizes, int n_in,
                              void* d_out, int out_size, void* d_ws, size_t ws_size,
                              hipStream_t stream) {
    const float* h  = (const float*)d_in[0];
    const int*   ei = (const int*)d_in[1];   // [2, E]: src then dst
    // d_in[2] = batch (implied by structure: graph = node/64)
    const float* w1 = (const float*)d_in[3];
    const float* b1 = (const float*)d_in[4];
    const float* w2 = (const float*)d_in[5];
    const float* b2 = (const float*)d_in[6];

    float* out   = (float*)d_out;
    float* out_h = out;                                  // N*D
    float* ge    = out + (size_t)NN * DD;                // G*D
    float* pos   = ge + (size_t)NUM_GRAPHS * DD;         // G*D
    float* pen   = pos + (size_t)NUM_GRAPHS * DD;        // G

    fused_graph_kernel<<<NUM_GRAPHS, 128, 0, stream>>>(
        h, ei, ei + EE, w1, b1, w2, b2, out_h, ge, pos, pen);

    (void)in_sizes; (void)n_in; (void)out_size; (void)d_ws; (void)ws_size;
}
